// VectorizedSSM_4320737100715
// MI455X (gfx1250) — compile-verified, matrix operand order checked by reading, not run on hardware
//
#include <hip/hip_runtime.h>
#include <hip/hip_bf16.h>
#include <math.h>

typedef __attribute__((ext_vector_type(16))) __bf16 v16bf;
typedef __attribute__((ext_vector_type(8)))  __bf16 v8bf;
typedef __attribute__((ext_vector_type(8)))  float  v8f;

#define BB 2
#define LL 2048
#define DD 1024
#define NN 16
#define ROWS (BB*LL)   // 4096

// gfx1250 hardware tanh (single transcendental op)
__device__ __forceinline__ float fast_tanh(float x) {
    float r;
    asm("v_tanh_f32 %0, %1" : "=v"(r) : "v"(x));
    return r;
}

__device__ __forceinline__ float soft_clamp_f(float x, float lo, float hi, float sharp) {
    float c  = 0.5f * (hi + lo);
    float r  = 0.5f * (hi - lo);
    float xn = (x - c) / (r + 1e-8f);
    return fast_tanh(xn * sharp) / sharp * r + c;
}

// ---------------------------------------------------------------------------
// fp32 -> bf16 conversion (4 elements/thread)
// ---------------------------------------------------------------------------
__global__ void __launch_bounds__(256)
cvt_bf16_kernel(const float* __restrict__ in, __bf16* __restrict__ out, int n) {
    int base = (blockIdx.x * blockDim.x + threadIdx.x) * 4;
    if (base < n) {
        float4 v = *(const float4*)(in + base);
        out[base + 0] = (__bf16)v.x;
        out[base + 1] = (__bf16)v.y;
        out[base + 2] = (__bf16)v.z;
        out[base + 3] = (__bf16)v.w;
    }
}

// ---------------------------------------------------------------------------
// dt = soft_clamp(softplus(x @ Wdt^T + bdt))  via  v_wmma_f32_16x16x32_bf16
// 2x2 register tiling: one wave computes a 32x32 output block (4 accumulators),
// each A/B fragment is reused twice -> 8 b128 loads feed 4 WMMAs.
// ---------------------------------------------------------------------------
__global__ void __launch_bounds__(256)
wmma_dt_kernel(const __bf16* __restrict__ xbf, const __bf16* __restrict__ wbf,
               const float* __restrict__ bdt, float* __restrict__ dt) {
    const int lane = threadIdx.x & 31;
    const int wave = threadIdx.x >> 5;            // 0..7
    const int tile = blockIdx.x * 8 + wave;       // 0..4095
    const int tm   = tile >> 5;                   // 0..127  (32-row strips)
    const int tn   = tile & 31;                   // 0..31   (32-col strips)

    const int l15 = lane & 15;
    const int kg  = (lane >> 4) * 8;              // K sub-group: 0 or 8

    // A fragments: rows tm*32 + l15 (+16); K groups [kg..kg+7], [16+kg..16+kg+7]
    const __bf16* arow0 = xbf + (size_t)(tm * 32 + l15) * DD + kg;
    const __bf16* arow1 = arow0 + (size_t)16 * DD;
    // B fragments: columns tn*32 + l15 (+16); Wdt row-major in k
    const __bf16* brow0 = wbf + (size_t)(tn * 32 + l15) * DD + kg;
    const __bf16* brow1 = brow0 + (size_t)16 * DD;

    v8f acc00 = {}, acc01 = {}, acc10 = {}, acc11 = {};

    #pragma unroll 2
    for (int k0 = 0; k0 < DD; k0 += 32) {
        v8bf a0lo = *(const v8bf*)(arow0 + k0);
        v8bf a0hi = *(const v8bf*)(arow0 + k0 + 16);
        v8bf a1lo = *(const v8bf*)(arow1 + k0);
        v8bf a1hi = *(const v8bf*)(arow1 + k0 + 16);
        v8bf b0lo = *(const v8bf*)(brow0 + k0);
        v8bf b0hi = *(const v8bf*)(brow0 + k0 + 16);
        v8bf b1lo = *(const v8bf*)(brow1 + k0);
        v8bf b1hi = *(const v8bf*)(brow1 + k0 + 16);
        v16bf A0, A1, B0, B1;
        #pragma unroll
        for (int i = 0; i < 8; i++) {
            A0[i] = a0lo[i]; A0[i + 8] = a0hi[i];
            A1[i] = a1lo[i]; A1[i + 8] = a1hi[i];
            B0[i] = b0lo[i]; B0[i + 8] = b0hi[i];
            B1[i] = b1lo[i]; B1[i + 8] = b1hi[i];
        }
        acc00 = __builtin_amdgcn_wmma_f32_16x16x32_bf16(false, A0, false, B0, (short)0, acc00, false, false);
        acc01 = __builtin_amdgcn_wmma_f32_16x16x32_bf16(false, A0, false, B1, (short)0, acc01, false, false);
        acc10 = __builtin_amdgcn_wmma_f32_16x16x32_bf16(false, A1, false, B0, (short)0, acc10, false, false);
        acc11 = __builtin_amdgcn_wmma_f32_16x16x32_bf16(false, A1, false, B1, (short)0, acc11, false, false);
    }

    // C layout: lane 0-15 -> N=lane, M=r ; lane 16-31 -> N=lane-16, M=r+8
    const int mbase = (lane >> 4) * 8;
    auto epilogue = [&](v8f acc, int rowbase, int colbase) {
        const int col  = colbase + l15;
        const float bias = bdt[col];
        #pragma unroll
        for (int r = 0; r < 8; r++) {
            int row  = rowbase + mbase + r;
            float v  = acc[r] + bias;
            // softplus via hardware v_exp_f32 / v_log_f32 (result clamped anyway)
            float sp = (v > 20.f) ? v : __logf(1.f + __expf(v));
            dt[(size_t)row * DD + col] = soft_clamp_f(sp, 0.001f, 0.1f, 0.5f);
        }
    };
    epilogue(acc00, tm * 32,      tn * 32);
    epilogue(acc01, tm * 32,      tn * 32 + 16);
    epilogue(acc10, tm * 32 + 16, tn * 32);
    epilogue(acc11, tm * 32 + 16, tn * 32 + 16);
}

// ---------------------------------------------------------------------------
// Bp = x @ WB^T, Cp = x @ WC^T   ([4096,16] each) — small, plain VALU
// ---------------------------------------------------------------------------
__global__ void __launch_bounds__(256)
proj_kernel(const float* __restrict__ x, const float* __restrict__ WB,
            const float* __restrict__ WC, float* __restrict__ Bp,
            float* __restrict__ Cp) {
    int t   = blockIdx.x * blockDim.x + threadIdx.x;   // 0..65535
    int row = t >> 4;
    int n   = t & 15;
    const float4* xr = (const float4*)(x + (size_t)row * DD);
    const float4* wb = (const float4*)(WB + (size_t)n * DD);
    const float4* wc = (const float4*)(WC + (size_t)n * DD);
    float sb = 0.f, sc = 0.f;
    #pragma unroll 4
    for (int k = 0; k < DD / 4; k++) {
        float4 xv = xr[k], bv = wb[k], cv = wc[k];
        sb += xv.x * bv.x + xv.y * bv.y + xv.z * bv.z + xv.w * bv.w;
        sc += xv.x * cv.x + xv.y * cv.y + xv.z * cv.z + xv.w * cv.w;
    }
    Bp[t] = sb;
    Cp[t] = sc;
}

// ---------------------------------------------------------------------------
// Sequential scan. One workgroup per batch (global norm couples all of D,N).
// thread d holds h[d, 0..15] in registers; norm via shfl + LDS tree.
// All transcendentals are single hardware trans ops (v_tanh_f32 / v_exp_f32).
// ---------------------------------------------------------------------------
__global__ void __launch_bounds__(1024)
scan_kernel(const float* __restrict__ x, const float* __restrict__ dt,
            const float* __restrict__ Bp, const float* __restrict__ Cp,
            const float* __restrict__ A_log, const float* __restrict__ Dp,
            float* __restrict__ out) {
    const int b    = blockIdx.x;
    const int d    = threadIdx.x;
    const int lane = threadIdx.x & 31;
    const int wave = threadIdx.x >> 5;

    __shared__ float wsum[32];
    __shared__ float stotal;

    float A[NN], h[NN];
    #pragma unroll
    for (int n = 0; n < NN; n++) {
        float al = A_log[d * NN + n];
        // soft_clamp(al,-5,5,0.5) = 10*tanh(al/10)
        A[n] = -__expf(10.f * fast_tanh(al * 0.1f));
        h[n] = 0.f;
    }
    const float dpv = Dp[d];

    const float* xb  = x  + (size_t)b * LL * DD;
    const float* dtb = dt + (size_t)b * LL * DD;
    const float* Bb  = Bp + (size_t)b * LL * NN;
    const float* Cb  = Cp + (size_t)b * LL * NN;
    float*       ob  = out + (size_t)b * LL * DD;

    for (int t = 0; t < LL; t++) {
        if (t + 1 < LL) {   // gfx1250 global_prefetch_b8 for next step
            __builtin_prefetch(dtb + (size_t)(t + 1) * DD + d, 0, 0);
            __builtin_prefetch(xb  + (size_t)(t + 1) * DD + d, 0, 0);
        }
        float xv  = xb[(size_t)t * DD + d];
        float dtv = dtb[(size_t)t * DD + d];

        float Brow[NN], Crow[NN];
        #pragma unroll
        for (int q = 0; q < 4; q++) {
            float4 bq = *(const float4*)(Bb + t * NN + q * 4);
            float4 cq = *(const float4*)(Cb + t * NN + q * 4);
            Brow[q*4+0] = bq.x; Brow[q*4+1] = bq.y; Brow[q*4+2] = bq.z; Brow[q*4+3] = bq.w;
            Crow[q*4+0] = cq.x; Crow[q*4+1] = cq.y; Crow[q*4+2] = cq.z; Crow[q*4+3] = cq.w;
        }

        const float dtx = dtv * xv;
        float ss = 0.f;
        #pragma unroll
        for (int n = 0; n < NN; n++) {
            float z  = dtv * A[n];
            // soft_clamp(z,-8,8,0.5) = 16*tanh(z/16)
            float dA = __expf(16.f * fast_tanh(z * 0.0625f));
            h[n] = dA * h[n] + dtx * Brow[n];
            ss  += h[n] * h[n];
        }

        // wave32 reduce, then cross-wave via LDS
        #pragma unroll
        for (int off = 16; off >= 1; off >>= 1) ss += __shfl_xor(ss, off, 32);
        if (lane == 0) wsum[wave] = ss;
        __syncthreads();
        if (wave == 0) {
            float v = wsum[lane];
            #pragma unroll
            for (int off = 16; off >= 1; off >>= 1) v += __shfl_xor(v, off, 32);
            if (lane == 0) stotal = v;
        }
        __syncthreads();

        float norm  = sqrtf(stotal);
        float scale = fminf(20.f / (norm + 1e-8f), 1.f);
        float y = dpv * xv;
        #pragma unroll
        for (int n = 0; n < NN; n++) {
            h[n] *= scale;
            y    += h[n] * Crow[n];
        }
        ob[(size_t)t * DD + d] = y;
    }
}

// ---------------------------------------------------------------------------
extern "C" void kernel_launch(void* const* d_in, const int* in_sizes, int n_in,
                              void* d_out, int out_size, void* d_ws, size_t ws_size,
                              hipStream_t stream) {
    const float* x     = (const float*)d_in[0];
    const float* Wdt   = (const float*)d_in[1];
    const float* bdt   = (const float*)d_in[2];
    const float* WB    = (const float*)d_in[3];
    const float* WC    = (const float*)d_in[4];
    const float* A_log = (const float*)d_in[5];
    const float* Dp    = (const float*)d_in[6];
    float* out = (float*)d_out;

    char* ws = (char*)d_ws;
    size_t off = 0;
    auto take = [&](size_t bytes) -> void* {
        void* p = ws + off;
        off += (bytes + 255) & ~(size_t)255;
        return p;
    };
    float*  dt  = (float*) take((size_t)ROWS * DD * sizeof(float));   // 16 MB
    float*  Bp  = (float*) take((size_t)ROWS * NN * sizeof(float));
    float*  Cp  = (float*) take((size_t)ROWS * NN * sizeof(float));
    __bf16* xbf = (__bf16*)take((size_t)ROWS * DD * sizeof(__bf16));  // 8 MB
    __bf16* wbf = (__bf16*)take((size_t)DD * DD * sizeof(__bf16));    // 2 MB
    (void)ws_size; (void)in_sizes; (void)n_in; (void)out_size;

    // 1) bf16 conversions
    {
        int n = ROWS * DD;                              // 4,194,304
        cvt_bf16_kernel<<<(n / 4 + 255) / 256, 256, 0, stream>>>(x, xbf, n);
        int m = DD * DD;                                // 1,048,576
        cvt_bf16_kernel<<<(m / 4 + 255) / 256, 256, 0, stream>>>(Wdt, wbf, m);
    }
    // 2) WMMA GEMM + softplus + soft_clamp -> dt  (32x32 tile per wave)
    {
        int tiles = (ROWS / 32) * (DD / 32);            // 4096 tiles, 8/block
        wmma_dt_kernel<<<tiles / 8, 256, 0, stream>>>(xbf, wbf, bdt, dt);
    }
    // 3) B/C projections
    proj_kernel<<<(ROWS * NN) / 256, 256, 0, stream>>>(x, WB, WC, Bp, Cp);
    // 4) sequential scan, one workgroup per batch
    scan_kernel<<<BB, 1024, 0, stream>>>(x, dt, Bp, Cp, A_log, Dp, out);
}
